// GCNWithAgg_35742717837379
// MI455X (gfx1250) — matrix operand order repaired and checked by reading, not compiled, not run on hardware
//
#include <hip/hip_runtime.h>

#define NODES 3072
#define INCH  128
#define HIDD  256
#define OUTD  32
#define HEADS 12

typedef __attribute__((ext_vector_type(16))) __bf16 v16bf;
typedef __attribute__((ext_vector_type(8)))  __bf16 v8bf;
typedef __attribute__((ext_vector_type(8)))  float  v8f;

static __device__ __forceinline__ v16bf combine16(v8bf lo, v8bf hi) {
    v16bf r;
#pragma unroll
    for (int e = 0; e < 8; ++e) { r[e] = lo[e]; r[e + 8] = hi[e]; }
    return r;
}

static __device__ __forceinline__ v8f wmma_bf16(v16bf a, v16bf b, v8f c) {
    return __builtin_amdgcn_wmma_f32_16x16x32_bf16(
        /*neg_a=*/false, a, /*neg_b=*/false, b,
        /*c_mod=*/(short)0, c, /*reuse_a=*/false, /*reuse_b=*/false);
}

// gfx1250 direct global->LDS async copy, 16B per lane. Tracked by ASYNCcnt.
static __device__ __forceinline__ void async_load_b128(const void* gptr, void* lptr) {
    unsigned lds_addr = (unsigned)(uintptr_t)lptr;  // LDS offset lives in addr[31:0]
    asm volatile("global_load_async_to_lds_b128 %0, %1, off"
                 :: "v"(lds_addr), "v"(gptr)
                 : "memory");
}

// ---------------------------------------------------------------------------
// init: zero degree counters + pooled vector (workspace is poisoned by harness)
// ---------------------------------------------------------------------------
__global__ void init_kernel(unsigned int* __restrict__ deg, float* __restrict__ g) {
    int id = blockIdx.x * blockDim.x + threadIdx.x;
    if (id < NODES) deg[id] = 0u;
    if (id < HIDD)  g[id] = 0.f;
}

// ---------------------------------------------------------------------------
// Fused: agg = sum_c w[c]*attn[c] + b ; A_hat (threshold + self loop);
// write A_hat^T as exact bf16 {0,1}; accumulate in-degree deg[t].
// ---------------------------------------------------------------------------
__global__ __launch_bounds__(256) void agg_kernel(
    const float* __restrict__ attn, const float* __restrict__ agg_w,
    const float* __restrict__ agg_b, __bf16* __restrict__ AT,
    unsigned int* __restrict__ deg)
{
    __shared__ float tile[32][33];
    __shared__ float wsh[HEADS];
    int tid = threadIdx.x;
    if (tid < HEADS) wsh[tid] = agg_w[tid];
    __syncthreads();

    int s0 = blockIdx.x * 32, t0 = blockIdx.y * 32;
    float b = agg_b[0];

    int tl = tid & 31, sl4 = (tid >> 5) * 4;
#pragma unroll
    for (int i = 0; i < 4; ++i) {
        int s = s0 + sl4 + i, t = t0 + tl;
        size_t base = (size_t)s * NODES + t;
        float v = b;
#pragma unroll
        for (int c = 0; c < HEADS; ++c)
            v += wsh[c] * attn[(size_t)c * NODES * NODES + base];
        float ah = (s == t) ? 1.f : (v > 0.f ? 1.f : 0.f);
        tile[sl4 + i][tl] = ah;
    }
    __syncthreads();

    int sl = tid & 31, tl4 = (tid >> 5) * 4;
#pragma unroll
    for (int i = 0; i < 4; ++i) {
        int t = tl4 + i;
        float ah = tile[sl][t];
        AT[(size_t)(t0 + t) * NODES + s0 + sl] = (__bf16)ah;  // exact 0/1
        unsigned long long m = __ballot(ah > 0.5f);           // wave32: low 32 bits
        if (sl == 0) atomicAdd(&deg[t0 + t], (unsigned)__popcll(m));
    }
}

__global__ void dinv_kernel(const unsigned int* __restrict__ deg, float* __restrict__ dinv) {
    int i = blockIdx.x * blockDim.x + threadIdx.x;
    if (i < NODES) {
        unsigned d = deg[i];
        dinv[i] = d ? rsqrtf((float)d) : 0.f;
    }
}

// ---------------------------------------------------------------------------
// f32 -> bf16 elementwise, and f32 -> bf16 transpose (for B^T operands)
// ---------------------------------------------------------------------------
__global__ void cvt_bf16_kernel(const float* __restrict__ in, __bf16* __restrict__ out, int n) {
    int id = blockIdx.x * blockDim.x + threadIdx.x;
    if (id < n) out[id] = (__bf16)in[id];
}

__global__ void transpose_bf16_kernel(const float* __restrict__ in, __bf16* __restrict__ out,
                                      int R, int C) {
    int id = blockIdx.x * blockDim.x + threadIdx.x;
    if (id >= R * C) return;
    int r = id % R, c = id / R;            // out is [C][R], contiguous in id
    out[(size_t)c * R + r] = (__bf16)in[(size_t)r * C + c];
}

// ---------------------------------------------------------------------------
// WMMA bf16 GEMM:  C[M, 256] = A[M, K] * B[K, 256], B given transposed (BT[256][K]).
// Block tile 64(M) x 128(N), 8 waves (2x4), wave tile 32x32, K-step 32.
// Double-buffered LDS staged by global_load_async_to_lds_b128 (ASYNCcnt).
// Epilogue MODE: 0 -> bf16 Out[n][m] = acc*dinv[m]           (transposed, scaled)
//               1 -> bf16 Out[m][n] = relu(acc*dinv[m]+bias[n])
//               2 -> f32  Out[m][n] = acc*dinv[m]+bias[n]
// ---------------------------------------------------------------------------
template <int MODE>
__global__ __launch_bounds__(256) void gemm_bf16_kernel(
    const __bf16* __restrict__ A, int lda,
    const __bf16* __restrict__ BT, int ldbt,
    void* __restrict__ Out, int ldo,
    const float* __restrict__ rowscale, const float* __restrict__ bias,
    int M, int K)
{
    constexpr int LSTR = 40;  // LDS row stride in halves: 80B, 16B aligned, conflict-free
    __shared__ __bf16 As[2][64 * LSTR];
    __shared__ __bf16 Bs[2][128 * LSTR];

    int tid = threadIdx.x, lane = tid & 31, wave = tid >> 5;
    int wm = (wave & 1) * 32;        // 2 waves along M
    int wn = (wave >> 1) * 32;       // 4 waves along N
    int m0 = blockIdx.x * 64, n0 = blockIdx.y * 128;

    v8f acc[2][2] = {};

    int srow = tid >> 2;             // 0..63
    int schunk = (tid & 3) * 8;      // halves, 16B aligned

    const __bf16* gA  = A  + (size_t)(m0 + srow) * lda  + schunk;
    const __bf16* gB0 = BT + (size_t)(n0 + srow) * ldbt + schunk;
    const __bf16* gB1 = BT + (size_t)(n0 + srow + 64) * ldbt + schunk;
    __bf16* lA  = &As[0][srow * LSTR + schunk];
    __bf16* lB0 = &Bs[0][srow * LSTR + schunk];
    __bf16* lB1 = &Bs[0][(srow + 64) * LSTR + schunk];

    const int nsteps = K / 32;

    // prologue: stage k-tile 0 into buffer 0
    async_load_b128(gA,  lA);
    async_load_b128(gB0, lB0);
    async_load_b128(gB1, lB1);

    for (int s = 0; s < nsteps; ++s) {
        int cur = s & 1;
        bool have_next = (s + 1) < nsteps;
        if (have_next) {
            int kn = (s + 1) * 32;
            int nb = cur ^ 1;
            async_load_b128(gA + kn,  lA  + nb * (64 * LSTR));
            async_load_b128(gB0 + kn, lB0 + nb * (128 * LSTR));
            async_load_b128(gB1 + kn, lB1 + nb * (128 * LSTR));
            asm volatile("s_wait_asynccnt 0x3" ::: "memory");  // current tile landed
        } else {
            asm volatile("s_wait_asynccnt 0x0" ::: "memory");
        }
        __syncthreads();  // current tile visible block-wide

        const __bf16* as = As[cur];
        const __bf16* bs = Bs[cur];

        // A fragments: lane = M row; lanes 0-15 K{0..7,16..23}, 16-31 K{8..15,24..31}
        v16bf af[2], bfr[2];
        {
            int row = lane & 15, kb = (lane >> 4) * 8;
#pragma unroll
            for (int i = 0; i < 2; ++i) {
                const __bf16* p = as + (wm + i * 16 + row) * LSTR + kb;
                af[i] = combine16(*(const v8bf*)p, *(const v8bf*)(p + 16));
            }
            // B fragments from BT rows: lane = N col; lanes 0-15 K{0..7,16..23}, 16-31 K{8..15,24..31}
            int kb2 = (lane >> 4) * 8;
#pragma unroll
            for (int j = 0; j < 2; ++j) {
                const __bf16* p = bs + (wn + j * 16 + row) * LSTR + kb2;
                bfr[j] = combine16(*(const v8bf*)p, *(const v8bf*)(p + 16));
            }
        }
#pragma unroll
        for (int i = 0; i < 2; ++i)
#pragma unroll
            for (int j = 0; j < 2; ++j)
                acc[i][j] = wmma_bf16(af[i], bfr[j], acc[i][j]);
        __syncthreads();  // done reading cur before next iteration overwrites it
    }

    // epilogue: C layout — lane L, vgpr r: M = r + (L>>4)*8, N = L&15
#pragma unroll
    for (int i = 0; i < 2; ++i) {
#pragma unroll
        for (int j = 0; j < 2; ++j) {
            int mbase = m0 + wm + i * 16 + (lane >> 4) * 8;
            int ncol  = n0 + wn + j * 16 + (lane & 15);
#pragma unroll
            for (int r = 0; r < 8; ++r) {
                int m = mbase + r;
                float v = acc[i][j][r] * rowscale[m];
                if (MODE == 0) {
                    ((__bf16*)Out)[(size_t)ncol * ldo + m] = (__bf16)v;
                } else if (MODE == 1) {
                    v += bias[ncol];
                    v = v > 0.f ? v : 0.f;
                    ((__bf16*)Out)[(size_t)m * ldo + ncol] = (__bf16)v;
                } else {
                    v += bias[ncol];
                    ((float*)Out)[(size_t)m * ldo + ncol] = v;
                }
            }
        }
    }
}

// ---------------------------------------------------------------------------
// Column mean pool (partial sums via atomics), then head MLP in one block.
// ---------------------------------------------------------------------------
__global__ __launch_bounds__(256) void pool_kernel(const float* __restrict__ H2,
                                                   float* __restrict__ g) {
    int d = threadIdx.x;
    int t0 = blockIdx.x * 128;
    float s = 0.f;
    for (int i = 0; i < 128; ++i) s += H2[(size_t)(t0 + i) * HIDD + d];
    atomicAdd(&g[d], s);
}

__global__ __launch_bounds__(256) void head_kernel(
    const float* __restrict__ g,
    const float* __restrict__ l1W, const float* __restrict__ l1b,
    const float* __restrict__ lng, const float* __restrict__ lnb,
    const float* __restrict__ l2W, const float* __restrict__ l2b,
    float* __restrict__ out)
{
    __shared__ float gs[HIDD], zs[HIDD], red[HIDD];
    int d = threadIdx.x;
    gs[d] = g[d] * (1.0f / NODES);
    __syncthreads();

    float acc = l1b[d];
    for (int k = 0; k < HIDD; ++k) acc += gs[k] * l1W[k * HIDD + d];
    float z = acc > 0.f ? acc : 0.f;

    red[d] = z; __syncthreads();
    for (int off = 128; off > 0; off >>= 1) {
        if (d < off) red[d] += red[d + off];
        __syncthreads();
    }
    float mu = red[0] * (1.0f / HIDD);
    __syncthreads();
    float dz = z - mu;
    red[d] = dz * dz; __syncthreads();
    for (int off = 128; off > 0; off >>= 1) {
        if (d < off) red[d] += red[d + off];
        __syncthreads();
    }
    float var = red[0] * (1.0f / HIDD);
    __syncthreads();
    zs[d] = dz * rsqrtf(var + 1e-5f) * lng[d] + lnb[d];
    __syncthreads();

    if (d < OUTD) {
        float o = l2b[d];
        for (int k = 0; k < HIDD; ++k) o += zs[k] * l2W[k * OUTD + d];
        out[d] = o;
    }
}

// ---------------------------------------------------------------------------
static inline size_t align256(size_t x) { return (x + 255) & ~(size_t)255; }

extern "C" void kernel_launch(void* const* d_in, const int* in_sizes, int n_in,
                              void* d_out, int out_size, void* d_ws, size_t ws_size,
                              hipStream_t stream) {
    const float* x      = (const float*)d_in[0];
    const float* attn   = (const float*)d_in[1];
    const float* agg_w  = (const float*)d_in[4];
    const float* agg_b  = (const float*)d_in[5];
    const float* W1     = (const float*)d_in[6];
    const float* b1     = (const float*)d_in[7];
    const float* W2     = (const float*)d_in[8];
    const float* b2     = (const float*)d_in[9];
    const float* l1W    = (const float*)d_in[10];
    const float* l1b    = (const float*)d_in[11];
    const float* lng    = (const float*)d_in[12];
    const float* lnb    = (const float*)d_in[13];
    const float* l2W    = (const float*)d_in[14];
    const float* l2b    = (const float*)d_in[15];
    float* out = (float*)d_out;

    // workspace carve-up
    char* p = (char*)d_ws;
    size_t off = 0;
    auto take = [&](size_t bytes) { void* q = p + off; off = align256(off + bytes); return q; };
    __bf16* AT     = (__bf16*)take((size_t)NODES * NODES * 2);  // Âᵀ, exact 0/1
    unsigned* deg  = (unsigned*)take(NODES * 4);
    float* dinv    = (float*)take(NODES * 4);
    __bf16* x_bf   = (__bf16*)take((size_t)NODES * INCH * 2);
    __bf16* W1T    = (__bf16*)take((size_t)HIDD * INCH * 2);    // [256][128]
    __bf16* W2T    = (__bf16*)take((size_t)HIDD * HIDD * 2);    // [256][256]
    __bf16* XW1sT  = (__bf16*)take((size_t)HIDD * NODES * 2);   // [256][3072]
    __bf16* H1     = (__bf16*)take((size_t)NODES * HIDD * 2);   // [3072][256]
    __bf16* XW2sT  = (__bf16*)take((size_t)HIDD * NODES * 2);   // [256][3072]
    float*  H2f    = (float*)take((size_t)NODES * HIDD * 4);    // [3072][256]
    float*  g      = (float*)take(HIDD * 4);
    (void)in_sizes; (void)n_in; (void)out_size; (void)ws_size;

    // 0) zero accumulators
    init_kernel<<<NODES / 256, 256, 0, stream>>>(deg, g);

    // 1) operand conversions (tiny)
    cvt_bf16_kernel<<<(NODES * INCH + 255) / 256, 256, 0, stream>>>(x, x_bf, NODES * INCH);
    transpose_bf16_kernel<<<(INCH * HIDD + 255) / 256, 256, 0, stream>>>(W1, W1T, INCH, HIDD);
    transpose_bf16_kernel<<<(HIDD * HIDD + 255) / 256, 256, 0, stream>>>(W2, W2T, HIDD, HIDD);

    // 2) fused attention-agg -> thresholded adjacency (452 MB stream, BW-bound)
    agg_kernel<<<dim3(NODES / 32, NODES / 32), 256, 0, stream>>>(attn, agg_w, agg_b, AT, deg);
    dinv_kernel<<<NODES / 256, 256, 0, stream>>>(deg, dinv);

    dim3 ggrid(NODES / 64, HIDD / 128);  // 48 x 2

    // 3) XW1s^T[d][s] = bf16( dinv[s] * (x @ W1)[s][d] )
    gemm_bf16_kernel<0><<<ggrid, 256, 0, stream>>>(x_bf, INCH, W1T, INCH,
                                                   XW1sT, NODES, dinv, nullptr, NODES, INCH);
    // 4) H1[t][d] = relu( dinv[t] * (Âᵀ @ XW1s)[t][d] + b1[d] )
    gemm_bf16_kernel<1><<<ggrid, 256, 0, stream>>>(AT, NODES, XW1sT, NODES,
                                                   H1, HIDD, dinv, b1, NODES, NODES);
    // 5) XW2s^T[d][s] = bf16( dinv[s] * (H1 @ W2)[s][d] )
    gemm_bf16_kernel<0><<<ggrid, 256, 0, stream>>>(H1, HIDD, W2T, HIDD,
                                                   XW2sT, NODES, dinv, nullptr, NODES, HIDD);
    // 6) H2[t][d] = dinv[t] * (Âᵀ @ XW2s)[t][d] + b2[d]   (f32)
    gemm_bf16_kernel<2><<<ggrid, 256, 0, stream>>>(AT, NODES, XW2sT, NODES,
                                                   H2f, HIDD, dinv, b2, NODES, NODES);

    // 7) mean pool + head MLP
    pool_kernel<<<NODES / 128, 256, 0, stream>>>(H2f, g);
    head_kernel<<<1, 256, 0, stream>>>(g, l1W, l1b, lng, lnb, l2W, l2b, out);
}
